// GraphAttentionLayer_78142634983461
// MI455X (gfx1250) — matrix operand order repaired and checked by reading, not compiled, not run on hardware
//
#include <hip/hip_runtime.h>
#include <hip/hip_bf16.h>
#include <math.h>

// Problem constants (match reference)
#define H_ 4
#define B_ 64
#define S_ 512
#define C_ 512
#define D_ 128
#define LN_EPS_ 1e-5f

typedef __bf16 bf16_t;
typedef bf16_t v16bf __attribute__((ext_vector_type(16)));
typedef bf16_t v8bf  __attribute__((ext_vector_type(8)));
typedef float  v8f   __attribute__((ext_vector_type(8)));
typedef float  v4f   __attribute__((ext_vector_type(4)));

// ---------------------------------------------------------------------------
// WMMA wrapper: D = A(16x32 bf16) * B(32x16 bf16) + C(16x16 f32)
// ---------------------------------------------------------------------------
__device__ __forceinline__ v8f wmma_bf16(v16bf a, v16bf b, v8f c) {
  return __builtin_amdgcn_wmma_f32_16x16x32_bf16(
      /*neg_a=*/false, a, /*neg_b=*/false, b,
      /*c_mod=*/(short)0, c, /*reuse_a=*/false, /*reuse_b=*/false);
}

__device__ __forceinline__ v16bf combine16(v8bf lo, v8bf hi) {
  v16bf r;
#pragma unroll
  for (int i = 0; i < 8; ++i) { r[i] = lo[i]; r[i + 8] = hi[i]; }
  return r;
}

// A fragment (16x32, row-major bf16 source). Per-lane: row = lane&15,
// elems 0-7 = K[k0+8*hh .. +7], elems 8-15 = K[k0+16+8*hh .. +7].
__device__ __forceinline__ v16bf load_a_bf16(const bf16_t* rowp, int k0, int hh) {
  v8bf lo = *(const v8bf*)(rowp + k0 + 8 * hh);
  v8bf hi = *(const v8bf*)(rowp + k0 + 16 + 8 * hh);
  return combine16(lo, hi);
}

// B fragment (32x16). colp = base of column n (K contiguous in memory).
// Per-lane: col = lane&15, elems 0-7 = K[k0+8*hh .. +7], elems 8-15 = K[k0+16+8*hh .. +7].
__device__ __forceinline__ v16bf load_b_bf16(const bf16_t* colp, int k0, int hh) {
  v8bf lo = *(const v8bf*)(colp + k0 + 8 * hh);
  v8bf hi = *(const v8bf*)(colp + k0 + 16 + 8 * hh);
  return combine16(lo, hi);
}

// A fragment from f32 source with in-register f32->bf16 conversion.
__device__ __forceinline__ v16bf load_a_f32cvt(const float* rowp, int k0, int hh) {
  v4f a0 = *(const v4f*)(rowp + k0 + 8 * hh);
  v4f a1 = *(const v4f*)(rowp + k0 + 8 * hh + 4);
  v4f a2 = *(const v4f*)(rowp + k0 + 16 + 8 * hh);
  v4f a3 = *(const v4f*)(rowp + k0 + 16 + 8 * hh + 4);
  v16bf r;
#pragma unroll
  for (int i = 0; i < 4; ++i) {
    r[i]      = (bf16_t)a0[i];
    r[4 + i]  = (bf16_t)a1[i];
    r[8 + i]  = (bf16_t)a2[i];
    r[12 + i] = (bf16_t)a3[i];
  }
  return r;
}

// ---------------------------------------------------------------------------
// k0: Wt[h][d][c] = bf16(W[h][c][d])   (makes B fragments K-contiguous)
// ---------------------------------------------------------------------------
__global__ __launch_bounds__(256) void k0_wt(const float* __restrict__ W,
                                             bf16_t* __restrict__ Wt) {
  int idx = blockIdx.x * 256 + threadIdx.x;     // H*C*D = 262144 threads
  int hd  = idx / (C_ * D_);
  int rem = idx % (C_ * D_);
  int c   = rem / D_;
  int d   = rem % D_;
  Wt[((size_t)hd * D_ + d) * C_ + c] = (bf16_t)W[idx];
}

// ---------------------------------------------------------------------------
// k1: projection hp[h][b][s][d] = h[b][s][:] @ W[h] + bias[h]
// grid = B*(S/16); block = 256 (8 waves). Wave w = d-tile w, all 4 heads.
// Also writes hpT[h][b][d][s] for the attn@hp GEMM.
// ---------------------------------------------------------------------------
__global__ __launch_bounds__(256) void k1_proj(const float* __restrict__ hsrc,
                                               const bf16_t* __restrict__ Wt,
                                               const float* __restrict__ bias,
                                               bf16_t* __restrict__ hp,
                                               bf16_t* __restrict__ hpT) {
  int wg   = blockIdx.x;
  int b    = wg >> 5;          // /32
  int sblk = wg & 31;
  int s0   = sblk * 16;
  int wave = threadIdx.x >> 5; // 0..7 = d-tile
  int lane = threadIdx.x & 31;
  int n    = lane & 15;
  int hh   = lane >> 4;
  int dcol = wave * 16 + n;

  const float* arow = hsrc + ((size_t)b * S_ + s0 + (lane & 15)) * C_;

  v8f acc[H_] = {};
  for (int k0 = 0; k0 < C_; k0 += 32) {
    v16bf a = load_a_f32cvt(arow, k0, hh);   // shared across heads
#pragma unroll
    for (int hd = 0; hd < H_; ++hd) {
      v16bf bb = load_b_bf16(Wt + ((size_t)(hd * D_ + dcol)) * C_, k0, hh);
      acc[hd] = wmma_bf16(a, bb, acc[hd]);
    }
  }

#pragma unroll
  for (int hd = 0; hd < H_; ++hd) {
    float  bv     = bias[hd * D_ + dcol];
    size_t hpbase = ((size_t)(hd * B_ + b) * S_ + s0) * D_ + dcol;
    v8bf pack;
#pragma unroll
    for (int r = 0; r < 8; ++r) {
      float  v  = acc[hd][r] + bv;          // row m = 8*hh + r
      bf16_t bw = (bf16_t)v;
      hp[hpbase + (size_t)(8 * hh + r) * D_] = bw;   // scattered b16
      pack[r] = bw;
    }
    // hpT[hd][b][dcol][s0 + 8*hh .. +7]  : contiguous 16B store
    size_t tbase = ((size_t)(hd * B_ + b) * D_ + dcol) * S_ + s0 + 8 * hh;
    *(v8bf*)(hpT + tbase) = pack;
  }
}

// ---------------------------------------------------------------------------
// k2: scores = (hp @ hp^T) * adj; softmax rows; attn bf16.
// grid = H*B*(S/16); block = 256. Wave w covers column tiles w, w+8, w+16, w+24.
// ---------------------------------------------------------------------------
__global__ __launch_bounds__(256) void k2_attn(const bf16_t* __restrict__ hp,
                                               const float* __restrict__ adj,
                                               bf16_t* __restrict__ attn) {
  __shared__ float sc[16][S_];   // 32 KB
  int wg   = blockIdx.x;
  int sblk = wg & 31;
  int hb   = wg >> 5;            // h*B + b
  int s0   = sblk * 16;
  int wave = threadIdx.x >> 5;
  int lane = threadIdx.x & 31;
  int n    = lane & 15;
  int hh   = lane >> 4;

  const bf16_t* hpb  = hp + (size_t)hb * S_ * D_;
  const bf16_t* arow = hpb + (size_t)(s0 + (lane & 15)) * D_;

  // Preload all A fragments (K = D = 128 -> 4 steps); reused for 4 col tiles.
  v16bf afr[4];
#pragma unroll
  for (int i = 0; i < 4; ++i) afr[i] = load_a_bf16(arow, 32 * i, hh);

#pragma unroll
  for (int j = 0; j < 4; ++j) {
    int ct   = wave + 8 * j;          // 0..31
    int tcol = ct * 16 + n;
    const bf16_t* colp = hpb + (size_t)tcol * D_;  // hp[t][:] is K-contiguous
    v8f acc = {};
#pragma unroll
    for (int i = 0; i < 4; ++i) {
      v16bf bb = load_b_bf16(colp, 32 * i, hh);
      acc = wmma_bf16(afr[i], bb, acc);
    }
#pragma unroll
    for (int r = 0; r < 8; ++r) {
      int m = 8 * hh + r;
      float av = adj[(size_t)(s0 + m) * S_ + tcol];
      sc[m][tcol] = acc[r] * av;      // multiplicative mask BEFORE softmax
    }
  }
  __syncthreads();

  // Softmax: 16 threads per row (within a half-wave -> shfl width 16).
  int   row  = threadIdx.x >> 4;      // 0..15
  int   part = threadIdx.x & 15;
  float mx = -3.402823466e38f;
  for (int cc = part; cc < S_; cc += 16) mx = fmaxf(mx, sc[row][cc]);
#pragma unroll
  for (int o = 8; o >= 1; o >>= 1) mx = fmaxf(mx, __shfl_xor(mx, o, 16));
  float sum = 0.0f;
  for (int cc = part; cc < S_; cc += 16) {
    float e = __expf(sc[row][cc] - mx);
    sc[row][cc] = e;
    sum += e;
  }
#pragma unroll
  for (int o = 8; o >= 1; o >>= 1) sum += __shfl_xor(sum, o, 16);
  float inv = 1.0f / sum;
  bf16_t* orow = attn + ((size_t)hb * S_ + s0 + row) * S_;
  for (int cc = part; cc < S_; cc += 16) orow[cc] = (bf16_t)(sc[row][cc] * inv);
}

// ---------------------------------------------------------------------------
// k3: out = attn @ hp (per head), concat heads, LayerNorm, exact GELU.
// grid = B*(S/16); block = 256. Wave w covers C-tiles w, w+8, w+16, w+24
// (tile ct -> head ct>>3, d-tile ct&7).
// ---------------------------------------------------------------------------
__global__ __launch_bounds__(256) void k3_out(const bf16_t* __restrict__ attn,
                                              const bf16_t* __restrict__ hpT,
                                              const float* __restrict__ gamma,
                                              const float* __restrict__ beta,
                                              float* __restrict__ out) {
  __shared__ float sc[16][C_];   // 32 KB
  int wg   = blockIdx.x;
  int b    = wg >> 5;
  int sblk = wg & 31;
  int s0   = sblk * 16;
  int wave = threadIdx.x >> 5;
  int lane = threadIdx.x & 31;
  int n    = lane & 15;
  int hh   = lane >> 4;

#pragma unroll
  for (int j = 0; j < 4; ++j) {
    int ct = wave + 8 * j;            // 0..31 over C tiles
    int hd = ct >> 3;
    int dt = ct & 7;
    const bf16_t* arow = attn + ((size_t)(hd * B_ + b) * S_ + s0 + (lane & 15)) * S_;
    const bf16_t* colp = hpT + ((size_t)(hd * B_ + b) * D_ + dt * 16 + n) * S_;
    v8f acc = {};
    for (int k0 = 0; k0 < S_; k0 += 32) {
      v16bf a  = load_a_bf16(arow, k0, hh);
      v16bf bb = load_b_bf16(colp, k0, hh);
      acc = wmma_bf16(a, bb, acc);
    }
#pragma unroll
    for (int r = 0; r < 8; ++r) sc[8 * hh + r][ct * 16 + n] = acc[r];
  }
  __syncthreads();

  // LayerNorm + GELU: 16 threads per row.
  int   row  = threadIdx.x >> 4;
  int   part = threadIdx.x & 15;
  float sum = 0.0f;
  for (int cc = part; cc < C_; cc += 16) sum += sc[row][cc];
#pragma unroll
  for (int o = 8; o >= 1; o >>= 1) sum += __shfl_xor(sum, o, 16);
  float mu = sum * (1.0f / (float)C_);
  float vs = 0.0f;
  for (int cc = part; cc < C_; cc += 16) {
    float d = sc[row][cc] - mu;
    vs += d * d;
  }
#pragma unroll
  for (int o = 8; o >= 1; o >>= 1) vs += __shfl_xor(vs, o, 16);
  float rstd = rsqrtf(vs * (1.0f / (float)C_) + LN_EPS_);
  float* orow = out + ((size_t)b * S_ + s0 + row) * C_;
  for (int cc = part; cc < C_; cc += 16) {
    float x = (sc[row][cc] - mu) * rstd * gamma[cc] + beta[cc];
    float g = 0.5f * x * (1.0f + erff(x * 0.70710678118654752440f));
    orow[cc] = g;
  }
}

// ---------------------------------------------------------------------------
// Launch
// ---------------------------------------------------------------------------
extern "C" void kernel_launch(void* const* d_in, const int* in_sizes, int n_in,
                              void* d_out, int out_size, void* d_ws, size_t ws_size,
                              hipStream_t stream) {
  (void)in_sizes; (void)n_in; (void)out_size; (void)ws_size;
  // setup_inputs order: t, h, W, b, adj, gamma, beta
  const float* h     = (const float*)d_in[1];
  const float* W     = (const float*)d_in[2];
  const float* bias  = (const float*)d_in[3];
  const float* adj   = (const float*)d_in[4];
  const float* gamma = (const float*)d_in[5];
  const float* beta  = (const float*)d_in[6];
  float* out = (float*)d_out;

  // Workspace layout (bytes):
  //   Wt   : H*D*C bf16      =    524,288
  //   hp   : H*B*S*D bf16    = 33,554,432
  //   hpT  : H*B*S*D bf16    = 33,554,432
  //   attn : H*B*S*S bf16    = 134,217,728
  char* ws = (char*)d_ws;
  bf16_t* Wt   = (bf16_t*)(ws);
  bf16_t* hp   = (bf16_t*)(ws + 524288);
  bf16_t* hpT  = (bf16_t*)(ws + 524288 + 33554432);
  bf16_t* attn = (bf16_t*)(ws + 524288 + 2 * 33554432);

  k0_wt  <<<(H_ * C_ * D_) / 256, 256, 0, stream>>>(W, Wt);
  k1_proj<<<B_ * (S_ / 16),       256, 0, stream>>>(h, Wt, bias, hp, hpT);
  k2_attn<<<H_ * B_ * (S_ / 16),  256, 0, stream>>>(hp, adj, attn);
  k3_out <<<B_ * (S_ / 16),       256, 0, stream>>>(attn, hpT, gamma, beta, out);
}
